// Difference_Module_27986006901188
// MI455X (gfx1250) — matrix operand order repaired and checked by reading, not compiled, not run on hardware
//
#include <hip/hip_runtime.h>
#include <hip/hip_bf16.h>

#define BATCH 8
#define SEQ   4096
#define DIMD  64
#define HID   256

typedef __attribute__((ext_vector_type(16))) __bf16 v16bf;
typedef __attribute__((ext_vector_type(8)))  __bf16 v8bf;
typedef __attribute__((ext_vector_type(2)))  __bf16 v2bf;
typedef __attribute__((ext_vector_type(8)))  float  v8f;

#define QK_SCALE   0.125f
#define LOG2E      1.4426950408889634f

#if defined(__AMDGCN__) && __has_builtin(__builtin_amdgcn_global_load_async_to_lds_b128)
#define ASYNC_STAGE 1
// exact parameter types per hipcc diagnostic: pointer to int __vector_size__(16)
typedef int v4i_vs __attribute__((vector_size(4 * sizeof(int))));
typedef __attribute__((address_space(1))) v4i_vs* as1_v4i_ptr;
typedef __attribute__((address_space(3))) v4i_vs* as3_v4i_ptr;
#else
#define ASYNC_STAGE 0
#endif

// Stage 16 bytes/lane from global into LDS.
__device__ __forceinline__ void stage_b128(const __bf16* g, __bf16* l) {
#if ASYNC_STAGE
  // gfx1250 async DMA: no VGPR round trip, tracked by ASYNCcnt
  __builtin_amdgcn_global_load_async_to_lds_b128(
      (as1_v4i_ptr)g, (as3_v4i_ptr)l, 0, 0);
#else
  *(v8bf*)l = *(const v8bf*)g;
#endif
}

__device__ __forceinline__ void stage_fence() {
#if ASYNC_STAGE
  asm volatile("s_wait_asynccnt 0x0" ::: "memory");
#endif
}

__device__ __forceinline__ v8f wmma_bf16(v16bf a, v16bf b, v8f c) {
  // D = A(16x32 bf16) * B(32x16 bf16) + C(16x16 f32)
  return __builtin_amdgcn_wmma_f32_16x16x32_bf16(
      false, a, false, b, (short)0, c, false, false);
}

// A-operand (16x32, row-major source, row stride in elements).
// Lane l<16 holds row l: K0..7 and K16..23; lane>=16 holds K8..15, K24..31.
__device__ __forceinline__ v16bf load_a_tile(const __bf16* base, int stride, int lane) {
  const int r = lane & 15, h = lane >> 4;
  const __bf16* p = base + r * stride + h * 8;
  v8bf lo = *(const v8bf*)(p);
  v8bf hi = *(const v8bf*)(p + 16);
  v16bf o;
#pragma unroll
  for (int i = 0; i < 8; ++i) { o[i] = lo[i]; o[i + 8] = hi[i]; }
  return o;
}

// B-operand from B^T row-major memory ("row" = output column, stride in elems).
// Lane l<16 holds col l: K0..7 and K16..23; lane>=16 holds K8..15, K24..31.
__device__ __forceinline__ v16bf load_bt_tile(const __bf16* base, int stride, int lane) {
  const int c = lane & 15, h = lane >> 4;
  const __bf16* p = base + c * stride + h * 8;
  v8bf lo = *(const v8bf*)(p);
  v8bf hi = *(const v8bf*)(p + 16);
  v16bf o;
#pragma unroll
  for (int i = 0; i < 8; ++i) { o[i] = lo[i]; o[i + 8] = hi[i]; }
  return o;
}

__device__ __forceinline__ float rsum16(float x) {   // layernorm epilogue only
#pragma unroll
  for (int m = 1; m < 16; m <<= 1) x += __shfl_xor(x, m, 32);
  return x;
}

__device__ __forceinline__ v16bf make_ones() {
  v16bf o;
#pragma unroll
  for (int i = 0; i < 16; ++i) o[i] = (__bf16)1.0f;
  return o;
}

// Intra-strip K permutation: score column m in [0,32) -> staging column.
// Lets each lane pack (p_S0, p_S1) as one b32 LDS store in the hot loops.
__device__ __forceinline__ int perm_n(int n) {
  return (n & ~31) + 2 * (n & 15) + ((n >> 4) & 1);
}

// ---------- conversion / transposition ----------
__global__ void k0_convert_qkv(const float* __restrict__ q, const float* __restrict__ k,
                               const float* __restrict__ v, __bf16* __restrict__ qbE,
                               __bf16* __restrict__ qbS, __bf16* __restrict__ kb,
                               __bf16* __restrict__ vT) {
  const int i = blockIdx.x * 256 + threadIdx.x;
  const int total = BATCH * SEQ * DIMD;
  if (i >= total) return;
  const float qi = q[i];
  qbE[i] = (__bf16)(qi * (QK_SCALE * LOG2E));   // K1: exp2(S) == exp(scale*q.k)
  qbS[i] = (__bf16)(qi * QK_SCALE);             // K2: S comes out pre-scaled
  kb[i]  = (__bf16)k[i];
  const int d = i % DIMD;
  const int n = (i / DIMD) % SEQ;
  const int b = i / (DIMD * SEQ);
  vT[((size_t)b * DIMD + d) * SEQ + perm_n(n)] = (__bf16)v[i];
}

__global__ void k0_convert_w(const float* __restrict__ Wd, const float* __restrict__ W1,
                             const float* __restrict__ W2, __bf16* __restrict__ WdT,
                             __bf16* __restrict__ W1T, __bf16* __restrict__ W2T) {
  const int i = blockIdx.x * 256 + threadIdx.x;
  if (i < DIMD * DIMD) {                       // W_dif [d_in][d_out] -> WdT [d_out][d_in]
    int r = i / DIMD, c = i % DIMD;
    WdT[c * DIMD + r] = (__bf16)Wd[i];
  }
  if (i < DIMD * HID) {
    int r1 = i / HID, c1 = i % HID;            // W1 [d][h] -> W1T [h][d]
    W1T[c1 * DIMD + r1] = (__bf16)W1[i];
    int r2 = i / DIMD, c2 = i % DIMD;          // W2 [h][d] -> W2T [d][h]
    W2T[c2 * HID + r2] = (__bf16)W2[i];
  }
}

// ---------- K1: attention (exp-sum via WMMA ones trick) + dif-proj -> v1^T ----------
__global__ __launch_bounds__(128) void k1_attn_dif(
    const float* __restrict__ vf, const __bf16* __restrict__ qbE,
    const __bf16* __restrict__ kb, const __bf16* __restrict__ vT,
    const __bf16* __restrict__ WdT, const float* __restrict__ bdif,
    __bf16* __restrict__ v1T) {
  __shared__ __align__(16) __bf16 kstg[2][32][64];   // double-buffered k strip (8KB)
  __shared__ __align__(16) __bf16 vstg[2][64][32];   // double-buffered v^T strip (8KB)
  __shared__ __align__(16) __bf16 pstg[4][16][32];   // per-wave P staging (4KB)
  __shared__ __align__(16) __bf16 tstg[4][16][64];   // per-wave (v-attn) staging (8KB)

  const int lane  = threadIdx.x & 31;
  const int wave  = threadIdx.x >> 5;
  const int b     = blockIdx.x >> 6;                 // SEQ/64 = 64 row tiles per batch
  const int n0    = ((blockIdx.x & 63) << 6) + (wave << 4);
  const int lc    = lane & 15;
  const int rbase = (lane >> 4) << 3;

  const __bf16* qrow = qbE + ((size_t)b * SEQ + n0) * DIMD;
  const v16bf aq0 = load_a_tile(qrow, DIMD, lane);
  const v16bf aq1 = load_a_tile(qrow + 32, DIMD, lane);
  const v16bf ones = make_ones();

  v8f O[4] = {};
  v8f Lacc = {};

  const __bf16* kbB = kb + (size_t)b * SEQ * DIMD;
  const __bf16* vTb = vT + (size_t)b * DIMD * SEQ;
  __bf16(*stgP)[32] = pstg[wave];

  const int t  = threadIdx.x;
  const int rr = t >> 1, cc = (t & 1) * 16;

  auto stage = [&](int buf, int j) {
    stage_b128(kbB + (size_t)j * DIMD + t * 16, &kstg[buf][0][0] + t * 16);
    stage_b128(vTb + (size_t)rr * SEQ + j + cc, &vstg[buf][rr][cc]);
  };

  stage(0, 0);
  for (int j0 = 0; j0 < SEQ; j0 += 32) {
    const int cur = (j0 >> 5) & 1;
    stage_fence();                          // async DMA into LDS complete
    __syncthreads();                        // staged strip visible; prev reads done
    if (j0 + 32 < SEQ) stage(cur ^ 1, j0 + 32);

    // batch all k B-tile loads, then interleave the 4 S WMMAs
    v16bf bk0a = load_bt_tile(&kstg[cur][0][0], 64, lane);
    v16bf bk1a = load_bt_tile(&kstg[cur][16][0], 64, lane);
    v16bf bk0b = load_bt_tile(&kstg[cur][0][0] + 32, 64, lane);
    v16bf bk1b = load_bt_tile(&kstg[cur][16][0] + 32, 64, lane);
    v8f S0 = {}, S1 = {};
    S0 = wmma_bf16(aq0, bk0a, S0);
    S1 = wmma_bf16(aq0, bk1a, S1);
    S0 = wmma_bf16(aq1, bk0b, S0);
    S1 = wmma_bf16(aq1, bk1b, S1);

    // issue the independent v^T B-tile loads; they drain during the exp block
    v16bf bv0 = load_bt_tile(&vstg[cur][0][0], 32, lane);
    v16bf bv1 = load_bt_tile(&vstg[cur][16][0], 32, lane);
    v16bf bv2 = load_bt_tile(&vstg[cur][32][0], 32, lane);
    v16bf bv3 = load_bt_tile(&vstg[cur][48][0], 32, lane);

    // P = exp2(S) (log2e*scale pre-folded into q); packed b32 stores via K-permute
#pragma unroll
    for (int r = 0; r < 8; ++r) {
      v2bf pk;
      pk[0] = (__bf16)__builtin_amdgcn_exp2f(S0[r]);
      pk[1] = (__bf16)__builtin_amdgcn_exp2f(S1[r]);
      *(v2bf*)(&stgP[rbase + r][2 * lc]) = pk;
    }
    v16bf ap = load_a_tile(&stgP[0][0], 32, lane);
    Lacc = wmma_bf16(ap, ones, Lacc);              // row-sums via matrix core
    O[0] = wmma_bf16(ap, bv0, O[0]);
    O[1] = wmma_bf16(ap, bv1, O[1]);
    O[2] = wmma_bf16(ap, bv2, O[2]);
    O[3] = wmma_bf16(ap, bv3, O[3]);
  }

  // attn = O / l; tv = v - attn, staged bf16 for the dif-projection
  const float* vrow = vf + ((size_t)b * SEQ + n0) * DIMD;
  __bf16(*stgT)[64] = tstg[wave];
#pragma unroll
  for (int r = 0; r < 8; ++r) {
    float inv = 1.0f / Lacc[r];                    // every col of Lacc == row sum
#pragma unroll
    for (int c = 0; c < 4; ++c) {
      float a = O[c][r] * inv;
      float tv = vrow[(rbase + r) * DIMD + (c * 16 + lc)] - a;
      stgT[rbase + r][c * 16 + lc] = (__bf16)tv;
    }
  }
  v16bf at0 = load_a_tile(&stgT[0][0], 64, lane);
  v16bf at1 = load_a_tile(&stgT[0][0] + 32, 64, lane);
#pragma unroll
  for (int c = 0; c < 4; ++c) {
    v8f acc = {};
    acc = wmma_bf16(at0, load_bt_tile(WdT + (size_t)(c * 16) * DIMD, DIMD, lane), acc);
    acc = wmma_bf16(at1, load_bt_tile(WdT + (size_t)(c * 16) * DIMD + 32, DIMD, lane), acc);
    float bd = bdif[c * 16 + lc];
    __bf16* v1row = v1T + ((size_t)b * DIMD + c * 16 + lc) * SEQ;
#pragma unroll
    for (int r = 0; r < 8; ++r)                    // K-permuted columns for K2's strips
      v1row[perm_n(n0 + rbase + r)] = (__bf16)(acc[r] + bd);
  }
}

// ---------- K2: v_new = scores@v1 + q; layernorm; MLP; out = mlp + v_new ----------
__global__ __launch_bounds__(128) void k2_vnew_mlp(
    const float* __restrict__ qf, const __bf16* __restrict__ qbS,
    const __bf16* __restrict__ kb, const __bf16* __restrict__ v1T,
    const float* __restrict__ gamma, const float* __restrict__ beta,
    const __bf16* __restrict__ W1T, const float* __restrict__ b1,
    const __bf16* __restrict__ W2T, const float* __restrict__ b2,
    float* __restrict__ out) {
  __shared__ __align__(16) __bf16 kstg[2][32][64];   // double-buffered k strip
  __shared__ __align__(16) __bf16 v1stg[2][64][32];  // double-buffered v1^T strip
  __shared__ __align__(16) __bf16 sstg[4][16][32];   // per-wave S staging
  __shared__ __align__(16) __bf16 hstg[4][16][64];   // per-wave h staging
  __shared__ __align__(16) __bf16 gstg[4][16][32];   // per-wave gelu staging

  const int lane  = threadIdx.x & 31;
  const int wave  = threadIdx.x >> 5;
  const int b     = blockIdx.x >> 6;
  const int n0    = ((blockIdx.x & 63) << 6) + (wave << 4);
  const int lc    = lane & 15;
  const int rbase = (lane >> 4) << 3;

  const __bf16* qrow = qbS + ((size_t)b * SEQ + n0) * DIMD;
  const v16bf aq0 = load_a_tile(qrow, DIMD, lane);
  const v16bf aq1 = load_a_tile(qrow + 32, DIMD, lane);

  const __bf16* kbB  = kb + (size_t)b * SEQ * DIMD;
  const __bf16* v1Tb = v1T + (size_t)b * DIMD * SEQ;
  __bf16(*stgS)[32] = sstg[wave];

  const int t  = threadIdx.x;
  const int rr = t >> 1, cc = (t & 1) * 16;

  auto stage = [&](int buf, int j) {
    stage_b128(kbB + (size_t)j * DIMD + t * 16, &kstg[buf][0][0] + t * 16);
    stage_b128(v1Tb + (size_t)rr * SEQ + j + cc, &v1stg[buf][rr][cc]);
  };

  v8f Vn[4] = {};
  stage(0, 0);
  for (int j0 = 0; j0 < SEQ; j0 += 32) {             // recompute scaled scores, S @ v1
    const int cur = (j0 >> 5) & 1;
    stage_fence();
    __syncthreads();
    if (j0 + 32 < SEQ) stage(cur ^ 1, j0 + 32);

    v16bf bk0a = load_bt_tile(&kstg[cur][0][0], 64, lane);
    v16bf bk1a = load_bt_tile(&kstg[cur][16][0], 64, lane);
    v16bf bk0b = load_bt_tile(&kstg[cur][0][0] + 32, 64, lane);
    v16bf bk1b = load_bt_tile(&kstg[cur][16][0] + 32, 64, lane);
    v8f S0 = {}, S1 = {};
    S0 = wmma_bf16(aq0, bk0a, S0);
    S1 = wmma_bf16(aq0, bk1a, S1);
    S0 = wmma_bf16(aq1, bk0b, S0);
    S1 = wmma_bf16(aq1, bk1b, S1);

    v16bf bv0 = load_bt_tile(&v1stg[cur][0][0], 32, lane);
    v16bf bv1 = load_bt_tile(&v1stg[cur][16][0], 32, lane);
    v16bf bv2 = load_bt_tile(&v1stg[cur][32][0], 32, lane);
    v16bf bv3 = load_bt_tile(&v1stg[cur][48][0], 32, lane);

#pragma unroll
    for (int r = 0; r < 8; ++r) {
      v2bf pk;
      pk[0] = (__bf16)S0[r];                         // scale pre-folded into q
      pk[1] = (__bf16)S1[r];
      *(v2bf*)(&stgS[rbase + r][2 * lc]) = pk;
    }
    v16bf as = load_a_tile(&stgS[0][0], 32, lane);
    Vn[0] = wmma_bf16(as, bv0, Vn[0]);
    Vn[1] = wmma_bf16(as, bv1, Vn[1]);
    Vn[2] = wmma_bf16(as, bv2, Vn[2]);
    Vn[3] = wmma_bf16(as, bv3, Vn[3]);
  }

  // + q residual
  const float* qrf = qf + ((size_t)b * SEQ + n0) * DIMD;
#pragma unroll
  for (int r = 0; r < 8; ++r)
#pragma unroll
    for (int c = 0; c < 4; ++c)
      Vn[c][r] += qrf[(rbase + r) * DIMD + c * 16 + lc];

  // layernorm -> h staged bf16 (one-time shuffle reductions)
  __bf16(*stgH)[64] = hstg[wave];
#pragma unroll
  for (int r = 0; r < 8; ++r) {
    float s = 0.f;
#pragma unroll
    for (int c = 0; c < 4; ++c) s += Vn[c][r];
    float mu = rsum16(s) * (1.0f / 64.0f);
    float s2 = 0.f;
#pragma unroll
    for (int c = 0; c < 4; ++c) { float d = Vn[c][r] - mu; s2 += d * d; }
    float rstd = rsqrtf(rsum16(s2) * (1.0f / 64.0f) + 1e-5f);
#pragma unroll
    for (int c = 0; c < 4; ++c) {
      float hh = (Vn[c][r] - mu) * rstd * gamma[c * 16 + lc] + beta[c * 16 + lc];
      stgH[rbase + r][c * 16 + lc] = (__bf16)hh;
    }
  }
  v16bf ah0 = load_a_tile(&stgH[0][0], 64, lane);
  v16bf ah1 = load_a_tile(&stgH[0][0] + 32, 64, lane);

  // MLP: gelu(h@W1+b1)@W2, K over HID in chunks of 32 (weights stay hot in L2)
  __bf16(*stgG)[32] = gstg[wave];
  v8f M[4] = {};
  for (int kk = 0; kk < 8; ++kk) {
    v16bf bw1_0a = load_bt_tile(W1T + (size_t)(kk * 2 + 0) * 16 * DIMD, DIMD, lane);
    v16bf bw1_1a = load_bt_tile(W1T + (size_t)(kk * 2 + 1) * 16 * DIMD, DIMD, lane);
    v16bf bw1_0b = load_bt_tile(W1T + (size_t)(kk * 2 + 0) * 16 * DIMD + 32, DIMD, lane);
    v16bf bw1_1b = load_bt_tile(W1T + (size_t)(kk * 2 + 1) * 16 * DIMD + 32, DIMD, lane);
    v8f u0 = {}, u1 = {};
    u0 = wmma_bf16(ah0, bw1_0a, u0);
    u1 = wmma_bf16(ah0, bw1_1a, u1);
    u0 = wmma_bf16(ah1, bw1_0b, u0);
    u1 = wmma_bf16(ah1, bw1_1b, u1);

    // W2 B-tiles are independent of the GELU work: issue loads first
    v16bf bw2_0 = load_bt_tile(W2T + (size_t)0  * HID + kk * 32, HID, lane);
    v16bf bw2_1 = load_bt_tile(W2T + (size_t)16 * HID + kk * 32, HID, lane);
    v16bf bw2_2 = load_bt_tile(W2T + (size_t)32 * HID + kk * 32, HID, lane);
    v16bf bw2_3 = load_bt_tile(W2T + (size_t)48 * HID + kk * 32, HID, lane);

    const float bb0 = b1[(kk * 2 + 0) * 16 + lc];
    const float bb1 = b1[(kk * 2 + 1) * 16 + lc];
#pragma unroll
    for (int r = 0; r < 8; ++r) {
      float x0 = u0[r] + bb0;
      float x1 = u1[r] + bb1;
      stgG[rbase + r][lc]      = (__bf16)(0.5f * x0 * (1.0f + erff(x0 * 0.70710678118654752f)));
      stgG[rbase + r][16 + lc] = (__bf16)(0.5f * x1 * (1.0f + erff(x1 * 0.70710678118654752f)));
    }
    v16bf ag = load_a_tile(&stgG[0][0], 32, lane);
    M[0] = wmma_bf16(ag, bw2_0, M[0]);
    M[1] = wmma_bf16(ag, bw2_1, M[1]);
    M[2] = wmma_bf16(ag, bw2_2, M[2]);
    M[3] = wmma_bf16(ag, bw2_3, M[3]);
  }

  float* orow = out + ((size_t)b * SEQ + n0) * DIMD;
#pragma unroll
  for (int c = 0; c < 4; ++c) {
    float bb = b2[c * 16 + lc];
#pragma unroll
    for (int r = 0; r < 8; ++r)
      orow[(rbase + r) * DIMD + c * 16 + lc] = M[c][r] + bb + Vn[c][r];
  }
}

extern "C" void kernel_launch(void* const* d_in, const int* in_sizes, int n_in,
                              void* d_out, int out_size, void* d_ws, size_t ws_size,
                              hipStream_t stream) {
  (void)in_sizes; (void)n_in; (void)out_size; (void)ws_size;
  const float* q  = (const float*)d_in[0];
  const float* k  = (const float*)d_in[1];
  const float* v  = (const float*)d_in[2];
  const float* Wd = (const float*)d_in[3];
  const float* bd = (const float*)d_in[4];
  const float* ga = (const float*)d_in[5];
  const float* be = (const float*)d_in[6];
  const float* W1 = (const float*)d_in[7];
  const float* b1 = (const float*)d_in[8];
  const float* W2 = (const float*)d_in[9];
  const float* b2 = (const float*)d_in[10];

  const size_t QKV = (size_t)BATCH * SEQ * DIMD;   // 2M elements
  char* w = (char*)d_ws;
  __bf16* qbE = (__bf16*)w; w += QKV * 2;
  __bf16* qbS = (__bf16*)w; w += QKV * 2;
  __bf16* kb  = (__bf16*)w; w += QKV * 2;
  __bf16* vT  = (__bf16*)w; w += QKV * 2;
  __bf16* v1T = (__bf16*)w; w += QKV * 2;
  __bf16* WdT = (__bf16*)w; w += (size_t)DIMD * DIMD * 2;
  __bf16* W1T = (__bf16*)w; w += (size_t)DIMD * HID * 2;
  __bf16* W2T = (__bf16*)w; w += (size_t)HID * DIMD * 2;

  hipLaunchKernelGGL(k0_convert_qkv, dim3((unsigned)(QKV / 256)), dim3(256), 0, stream,
                     q, k, v, qbE, qbS, kb, vT);
  hipLaunchKernelGGL(k0_convert_w, dim3(64), dim3(256), 0, stream,
                     Wd, W1, W2, WdT, W1T, W2T);
  hipLaunchKernelGGL(k1_attn_dif, dim3(BATCH * SEQ / 64), dim3(128), 0, stream,
                     v, qbE, kb, vT, WdT, bd, v1T);
  hipLaunchKernelGGL(k2_vnew_mlp, dim3(BATCH * SEQ / 64), dim3(128), 0, stream,
                     q, qbS, kb, v1T, ga, be, W1T, b1, W2T, b2, (float*)d_out);
}
